// ContextBlender_12438225289594
// MI455X (gfx1250) — compile-verified, matrix operand order checked by reading, not run on hardware
//
#include <hip/hip_runtime.h>
#include <hip/hip_bf16.h>

typedef __attribute__((ext_vector_type(16))) __bf16 v16bf;
typedef __attribute__((ext_vector_type(8)))  float  v8f;

#define BN 64
#define PP 100
#define CC 256
#define MM 2048
#define KVD 265
#define KVP 288          // KVD padded to multiple of 32
#define HH 8
#define DH 32
#define FFD 1024
#define NTOT (BN*PP)     // 6400
#define NMROWS (BN*MM)   // 131072

__device__ inline __bf16 f2bf(float f) {
  unsigned u = __builtin_bit_cast(unsigned, f);
  unsigned r = (u + 0x7FFFu + ((u >> 16) & 1u)) >> 16;
  return __builtin_bit_cast(__bf16, (unsigned short)r);
}

__device__ inline v8f zero8() {
  v8f z = {0.f,0.f,0.f,0.f,0.f,0.f,0.f,0.f};
  return z;
}

__device__ inline v8f wmma_bf16(v16bf a, v16bf b, v8f c) {
  return __builtin_amdgcn_wmma_f32_16x16x32_bf16(false, a, false, b, (short)0, c, false, false);
}

// Fragment load for the 16-bit WMMA A/B layout: lane's 16 bf16 elements are two
// contiguous 8-element runs at k0 + half*8 and k0 + 16 + half*8 -> two b128 loads.
__device__ inline v16bf load_frag(const __bf16* __restrict__ base, int stride_elems,
                                  int idx, int k0, int half) {
  const __bf16* p = base + (long)idx * stride_elems + k0 + half * 8;
  union { v16bf v; uint4 q[2]; } u;
  u.q[0] = *(const uint4*)(p);
  u.q[1] = *(const uint4*)(p + 16);
  return u.v;
}

// Same, reading a staged panel (LDS): panel holds [cols][32 k] bf16, col-major runs.
__device__ inline v16bf load_frag_lds(const __bf16* p) {
  union { v16bf v; uint4 q[2]; } u;
  u.q[0] = *(const uint4*)(p);
  u.q[1] = *(const uint4*)(p + 16);
  return u.v;
}

// CDNA5 async copy: 16B global -> LDS per lane, tracked by ASYNCcnt.
__device__ inline void async_b128(const __bf16* gsrc, const __bf16* ldst) {
  unsigned loff = (unsigned)(size_t)ldst;                 // low 32 bits = LDS offset
  asm volatile("global_load_async_to_lds_b128 %0, %1, off"
               :: "v"(loff), "v"((unsigned long long)gsrc)
               : "memory");
}

__device__ inline void wait_async0() {
  asm volatile("s_wait_asynccnt 0x0" ::: "memory");
}

// ---------------- pool: z[n,c] = mean_{7x7}; also bf16 copy ----------------
__global__ void cb_pool(const float* __restrict__ feat, float* __restrict__ z,
                        __bf16* __restrict__ zb) {
  int n = blockIdx.x, c = threadIdx.x;
  const float* p = feat + ((long)n * CC + c) * 49;
  float s = 0.f;
#pragma unroll
  for (int i = 0; i < 49; ++i) s += p[i];
  s *= (1.0f / 49.0f);
  z[(long)n * CC + c] = s;
  zb[(long)n * CC + c] = f2bf(s);
}

// ---------------- memory [131072,265] f32 -> padded bf16 [131072,288] ----------------
__global__ void cb_cvt_mem(const float* __restrict__ mem, __bf16* __restrict__ memb) {
  long row = blockIdx.x;
  int k = threadIdx.x;                       // blockDim = 288
  memb[row * KVP + k] = (k < KVD) ? f2bf(mem[row * KVD + k]) : f2bf(0.f);
}

// ---------------- weight [K,N] f32 -> transposed padded bf16 [N,Kpad] ----------------
__global__ void cb_cvt_wt(const float* __restrict__ W, __bf16* __restrict__ Wt,
                          int K, int N, int Kpad) {
  int n = blockIdx.x;
  int k = blockIdx.y * 256 + threadIdx.x;
  if (k >= Kpad) return;
  Wt[(long)n * Kpad + k] = (k < K) ? f2bf(W[(long)k * N + n]) : f2bf(0.f);
}

// ---------------- generic GEMM: out = act(A[M,Kpad]bf16 @ Bt[N,Kpad]^T + bias) ------
// 4-wave block: 128 rows x 64 cols; shared B panel staged in LDS via async copy,
// double-buffered (panel i+1 in flight while computing panel i).
__global__ void cb_gemm(const __bf16* __restrict__ A, const __bf16* __restrict__ Bt,
                        const float* __restrict__ bias,
                        float* __restrict__ out_f32, __bf16* __restrict__ out_bf16,
                        int Kpad, int Ncols, int relu) {
  __shared__ __bf16 panel[2][64 * 32];
  int tid = threadIdx.x;
  int lane = tid & 31, wv = tid >> 5;
  int half = lane >> 4, il = lane & 15;
  int row0 = blockIdx.x * 128 + wv * 32;
  int n0 = blockIdx.y * 64;

  // each thread copies 2x16B chunks of the 64x32 panel
  int c0 = tid >> 2, part0 = tid & 3;                 // chunk tid
  int c1 = (tid + 128) >> 2, part1 = tid & 3;         // chunk tid+128
  auto issue_panel = [&](int buf, int k0) {
    async_b128(Bt + (long)(n0 + c0) * Kpad + k0 + part0 * 8, &panel[buf][c0 * 32 + part0 * 8]);
    async_b128(Bt + (long)(n0 + c1) * Kpad + k0 + part1 * 8, &panel[buf][c1 * 32 + part1 * 8]);
  };

  v8f acc[8];
#pragma unroll
  for (int t = 0; t < 8; ++t) acc[t] = zero8();

  issue_panel(0, 0);
  int nk = Kpad >> 5;
  for (int i = 0; i < nk; ++i) {
    int k0 = i << 5;
    wait_async0();        // own async copies done
    __syncthreads();      // whole panel visible; all waves past previous reads
    if (i + 1 < nk) issue_panel((i + 1) & 1, k0 + 32);

    v16bf a0 = load_frag(A, Kpad, row0 + il, k0, half);
    v16bf a1 = load_frag(A, Kpad, row0 + 16 + il, k0, half);
    const __bf16* pp = &panel[i & 1][0];
#pragma unroll
    for (int t = 0; t < 4; ++t) {
      v16bf b = load_frag_lds(pp + (t * 16 + il) * 32 + half * 8);
      acc[t]     = wmma_bf16(a0, b, acc[t]);
      acc[4 + t] = wmma_bf16(a1, b, acc[4 + t]);
    }
  }
#pragma unroll
  for (int rt = 0; rt < 2; ++rt) {
#pragma unroll
    for (int t = 0; t < 4; ++t) {
      int col = n0 + t * 16 + il;
      float bv = bias[col];
#pragma unroll
      for (int e = 0; e < 8; ++e) {
        int row = row0 + rt * 16 + e + 8 * half;
        float v = acc[rt * 4 + t][e] + bv;
        if (relu) v = fmaxf(v, 0.0f);
        if (out_f32)  out_f32[(long)row * Ncols + col] = v;
        if (out_bf16) out_bf16[(long)row * Ncols + col] = f2bf(v);
      }
    }
  }
}

// ---------------- fused K/V projection ----------------
// 4-wave block: 128 rows x 32 cols of BOTH K and V; Wk/Wv panels async-staged in LDS.
// Kb: [B,H,M,32] bf16 ; Vtb: [B,H,32,M] bf16
__global__ void cb_kvproj(const __bf16* __restrict__ memb,
                          const __bf16* __restrict__ Wkt, const float* __restrict__ bk,
                          const __bf16* __restrict__ Wvt, const float* __restrict__ bv,
                          __bf16* __restrict__ Kb, __bf16* __restrict__ Vtb) {
  __shared__ __bf16 panelK[2][32 * 32];
  __shared__ __bf16 panelV[2][32 * 32];
  int tid = threadIdx.x;
  int lane = tid & 31, wv = tid >> 5;
  int half = lane >> 4, il = lane & 15;
  int row0 = blockIdx.x * 128 + wv * 32;   // global memory row (b*2048+m)
  int n0 = blockIdx.y * 32;                // output channel

  int c = tid >> 2, part = tid & 3;        // 32 cols * 4 chunks = 128 = blockDim
  auto issue_panels = [&](int buf, int k0) {
    async_b128(Wkt + (long)(n0 + c) * KVP + k0 + part * 8, &panelK[buf][c * 32 + part * 8]);
    async_b128(Wvt + (long)(n0 + c) * KVP + k0 + part * 8, &panelV[buf][c * 32 + part * 8]);
  };

  v8f kacc[4], vacc[4];
#pragma unroll
  for (int t = 0; t < 4; ++t) { kacc[t] = zero8(); vacc[t] = zero8(); }

  issue_panels(0, 0);
  const int nk = KVP >> 5;                  // 9
  for (int i = 0; i < nk; ++i) {
    int k0 = i << 5;
    wait_async0();
    __syncthreads();
    if (i + 1 < nk) issue_panels((i + 1) & 1, k0 + 32);

    v16bf a0 = load_frag(memb, KVP, row0 + il, k0, half);
    v16bf a1 = load_frag(memb, KVP, row0 + 16 + il, k0, half);
    const __bf16* pk = &panelK[i & 1][0];
    const __bf16* pv = &panelV[i & 1][0];
#pragma unroll
    for (int t = 0; t < 2; ++t) {
      v16bf bkf = load_frag_lds(pk + (t * 16 + il) * 32 + half * 8);
      kacc[t]     = wmma_bf16(a0, bkf, kacc[t]);
      kacc[2 + t] = wmma_bf16(a1, bkf, kacc[2 + t]);
      v16bf bvf = load_frag_lds(pv + (t * 16 + il) * 32 + half * 8);
      vacc[t]     = wmma_bf16(a0, bvf, vacc[t]);
      vacc[2 + t] = wmma_bf16(a1, bvf, vacc[2 + t]);
    }
  }
#pragma unroll
  for (int rt = 0; rt < 2; ++rt) {
#pragma unroll
    for (int t = 0; t < 2; ++t) {
      int col = n0 + t * 16 + il;
      int h = col >> 5, d = col & 31;
      float bkv = bk[col], bvv = bv[col];
#pragma unroll
      for (int e = 0; e < 8; ++e) {
        int gm = row0 + rt * 16 + e + 8 * half;
        int b = gm >> 11, m = gm & 2047;
        Kb [(((long)(b * HH + h)) * MM + m) * DH + d] = f2bf(kacc[rt * 2 + t][e] + bkv);
        Vtb[(((long)(b * HH + h)) * DH + d) * MM + m] = f2bf(vacc[rt * 2 + t][e] + bvv);
      }
    }
  }
}

// ---------------- flash attention per (b, h, p-tile), one wave ----------------
__global__ void cb_attn(const __bf16* __restrict__ q,   // [6400, 256]
                        const __bf16* __restrict__ Kb,  // [B,H,M,32]
                        const __bf16* __restrict__ Vtb, // [B,H,32,M]
                        __bf16* __restrict__ attnb,     // [6400, 256]
                        float scaler) {
  __shared__ __bf16 lds_p[16 * 32];
  int lane = threadIdx.x & 31;
  int pt = blockIdx.x, h = blockIdx.y, b = blockIdx.z;
  int p0 = pt * 16;
  int r = lane & 15, half = lane >> 4, il = lane & 15;

  int prow = p0 + r; if (prow > PP - 1) prow = PP - 1;   // clamp; masked on store
  v16bf qa = load_frag(q + ((long)(b * PP + prow)) * CC + h * DH, 0, 0, 0, half);

  const __bf16* Kbh = Kb  + ((long)(b * HH + h)) * MM * DH;
  const __bf16* Vth = Vtb + ((long)(b * HH + h)) * DH * MM;

  float mx[8], l[8];
#pragma unroll
  for (int e = 0; e < 8; ++e) { mx[e] = -1e30f; l[e] = 0.f; }
  v8f o0 = zero8(), o1 = zero8();

  for (int m0 = 0; m0 < MM; m0 += 32) {
    v16bf kb0 = load_frag(Kbh + (long)m0 * DH, DH, il, 0, half);
    v16bf kb1 = load_frag(Kbh + (long)m0 * DH, DH, 16 + il, 0, half);
    v8f s0 = wmma_bf16(qa, kb0, zero8());
    v8f s1 = wmma_bf16(qa, kb1, zero8());

    float alpha[8];
#pragma unroll
    for (int e = 0; e < 8; ++e) {
      float a0 = s0[e] * scaler, a1 = s1[e] * scaler;
      float cm = fmaxf(a0, a1);
#pragma unroll
      for (int off = 1; off < 16; off <<= 1) cm = fmaxf(cm, __shfl_xor(cm, off, 16));
      float nm = fmaxf(mx[e], cm);
      float al = __expf(mx[e] - nm);
      float e0 = __expf(a0 - nm), e1 = __expf(a1 - nm);
      float cs = e0 + e1;
#pragma unroll
      for (int off = 1; off < 16; off <<= 1) cs += __shfl_xor(cs, off, 16);
      l[e] = l[e] * al + cs;
      mx[e] = nm;
      alpha[e] = al;
      s0[e] = e0; s1[e] = e1;
    }
#pragma unroll
    for (int e = 0; e < 8; ++e) { o0[e] *= alpha[e]; o1[e] *= alpha[e]; }

    // transpose probs: D-layout -> A-layout via LDS (16x32 bf16)
    __syncthreads();
#pragma unroll
    for (int e = 0; e < 8; ++e) {
      int row = e + 8 * half;
      lds_p[row * 32 + il]      = f2bf(s0[e]);
      lds_p[row * 32 + 16 + il] = f2bf(s1[e]);
    }
    __syncthreads();
    v16bf pa = load_frag_lds(lds_p + r * 32 + half * 8);

    v16bf vb0 = load_frag(Vth + m0, MM, il, 0, half);
    v16bf vb1 = load_frag(Vth + m0, MM, 16 + il, 0, half);
    o0 = wmma_bf16(pa, vb0, o0);
    o1 = wmma_bf16(pa, vb1, o1);
  }

#pragma unroll
  for (int e = 0; e < 8; ++e) { float inv = 1.0f / l[e]; o0[e] *= inv; o1[e] *= inv; }
#pragma unroll
  for (int e = 0; e < 8; ++e) {
    int row = p0 + e + 8 * half;
    if (row < PP) {
      long o = ((long)(b * PP + row)) * CC + h * DH;
      attnb[o + il]      = f2bf(o0[e]);
      attnb[o + 16 + il] = f2bf(o1[e]);
    }
  }
}

// ---------------- residual + LayerNorm (in-place on z), wave per row ----------------
__global__ void cb_add_ln(float* __restrict__ z, const float* __restrict__ f,
                          const float* __restrict__ g, const float* __restrict__ be,
                          __bf16* __restrict__ zb) {
  int row = blockIdx.x;
  int lane = threadIdx.x & 31;
  float x[8], s1 = 0.f, s2 = 0.f;
  float* zr = z + (long)row * CC;
  const float* fr = f + (long)row * CC;
#pragma unroll
  for (int j = 0; j < 8; ++j) {
    int c = j * 32 + lane;
    float v = zr[c] + fr[c];
    x[j] = v; s1 += v; s2 += v * v;
  }
#pragma unroll
  for (int off = 1; off < 32; off <<= 1) { s1 += __shfl_xor(s1, off); s2 += __shfl_xor(s2, off); }
  float mean = s1 * (1.0f / CC);
  float var = s2 * (1.0f / CC) - mean * mean;
  float rinv = rsqrtf(var + 1e-5f);
  __bf16* zbr = zb + (long)row * CC;
#pragma unroll
  for (int j = 0; j < 8; ++j) {
    int c = j * 32 + lane;
    float y = (x[j] - mean) * rinv * g[c] + be[c];
    zr[c] = y;
    zbr[c] = f2bf(y);
  }
}

// ---------------- out = features + broadcast(zp) ----------------
__global__ void cb_bcast_add(const float* __restrict__ feat, const float* __restrict__ zp,
                             float* __restrict__ out) {
  long nc = (long)blockIdx.x * blockDim.x + threadIdx.x;
  if (nc >= (long)NTOT * CC) return;
  const float* f = feat + nc * 49;
  float* o = out + nc * 49;
  float v = zp[nc];
#pragma unroll
  for (int i = 0; i < 49; ++i) o[i] = f[i] + v;
}

extern "C" void kernel_launch(void* const* d_in, const int* in_sizes, int n_in,
                              void* d_out, int out_size, void* d_ws, size_t ws_size,
                              hipStream_t stream) {
  const float* feat = (const float*)d_in[0];
  const float* mem  = (const float*)d_in[1];
  const float* Wq = (const float*)d_in[2];  const float* bq = (const float*)d_in[3];
  const float* Wk = (const float*)d_in[4];  const float* bk = (const float*)d_in[5];
  const float* Wv = (const float*)d_in[6];  const float* bv = (const float*)d_in[7];
  const float* Wf = (const float*)d_in[8];  const float* bf = (const float*)d_in[9];
  const float* g1 = (const float*)d_in[10]; const float* be1 = (const float*)d_in[11];
  const float* g2 = (const float*)d_in[12]; const float* be2 = (const float*)d_in[13];
  const float* W1 = (const float*)d_in[14]; const float* b1 = (const float*)d_in[15];
  const float* W2 = (const float*)d_in[16]; const float* b2 = (const float*)d_in[17];
  const float* Wp1 = (const float*)d_in[18]; const float* bp1 = (const float*)d_in[19];
  const float* Wp2 = (const float*)d_in[20]; const float* bp2 = (const float*)d_in[21];
  float* out = (float*)d_out;

  char* ws = (char*)d_ws;
  size_t off = 0;
  auto carve = [&](size_t bytes) {
    char* p = ws + off;
    off = (off + bytes + 255) & ~(size_t)255;
    return p;
  };
  float*  z     = (float*) carve((size_t)NTOT * CC * 4);
  float*  tmp   = (float*) carve((size_t)NTOT * CC * 4);
  __bf16* zb    = (__bf16*)carve((size_t)NTOT * CC * 2);
  __bf16* attnb = (__bf16*)carve((size_t)NTOT * CC * 2);
  __bf16* qb    = (__bf16*)carve((size_t)NTOT * CC * 2);
  __bf16* h1b   = (__bf16*)carve((size_t)NTOT * FFD * 2);
  __bf16* Kb    = (__bf16*)carve((size_t)BN * HH * MM * DH * 2);
  __bf16* Vtb   = (__bf16*)carve((size_t)BN * HH * DH * MM * 2);
  __bf16* memb  = (__bf16*)carve((size_t)NMROWS * KVP * 2);
  __bf16* Wqt   = (__bf16*)carve((size_t)2 * CC * CC * 2);
  __bf16* Wkt   = (__bf16*)carve((size_t)2 * CC * KVP * 2);
  __bf16* Wvt   = (__bf16*)carve((size_t)2 * CC * KVP * 2);
  __bf16* Wft   = (__bf16*)carve((size_t)2 * CC * CC * 2);
  __bf16* W1t   = (__bf16*)carve((size_t)2 * FFD * CC * 2);
  __bf16* W2t   = (__bf16*)carve((size_t)2 * CC * FFD * 2);
  __bf16* Wp1t  = (__bf16*)carve((size_t)FFD * CC * 2);
  __bf16* Wp2t  = (__bf16*)carve((size_t)CC * FFD * 2);

  const float scaler = 0.17677669529663687f;  // 1/sqrt(32)

  cb_pool<<<dim3(NTOT), dim3(CC), 0, stream>>>(feat, z, zb);
  cb_cvt_mem<<<dim3(NMROWS), dim3(KVP), 0, stream>>>(mem, memb);

  for (int i = 0; i < 2; ++i) {
    cb_cvt_wt<<<dim3(CC, 1), dim3(256), 0, stream>>>(Wq + (long)i * CC * CC,  Wqt + (long)i * CC * CC,  CC,  CC,  CC);
    cb_cvt_wt<<<dim3(CC, 2), dim3(256), 0, stream>>>(Wk + (long)i * KVD * CC, Wkt + (long)i * CC * KVP, KVD, CC,  KVP);
    cb_cvt_wt<<<dim3(CC, 2), dim3(256), 0, stream>>>(Wv + (long)i * KVD * CC, Wvt + (long)i * CC * KVP, KVD, CC,  KVP);
    cb_cvt_wt<<<dim3(CC, 1), dim3(256), 0, stream>>>(Wf + (long)i * CC * CC,  Wft + (long)i * CC * CC,  CC,  CC,  CC);
    cb_cvt_wt<<<dim3(FFD, 1), dim3(256), 0, stream>>>(W1 + (long)i * CC * FFD, W1t + (long)i * FFD * CC, CC, FFD, CC);
    cb_cvt_wt<<<dim3(CC, 4), dim3(256), 0, stream>>>(W2 + (long)i * FFD * CC, W2t + (long)i * CC * FFD, FFD, CC, FFD);
  }
  cb_cvt_wt<<<dim3(FFD, 1), dim3(256), 0, stream>>>(Wp1, Wp1t, CC, FFD, CC);
  cb_cvt_wt<<<dim3(CC, 4), dim3(256), 0, stream>>>(Wp2, Wp2t, FFD, CC, FFD);

  for (int i = 0; i < 2; ++i) {
    cb_kvproj<<<dim3(NMROWS / 128, CC / 32), dim3(128), 0, stream>>>(
        memb, Wkt + (long)i * CC * KVP, bk + i * CC,
              Wvt + (long)i * CC * KVP, bv + i * CC, Kb, Vtb);
    // q = zb @ Wq + bq (bf16 out)
    cb_gemm<<<dim3(NTOT / 128, CC / 64), dim3(128), 0, stream>>>(
        zb, Wqt + (long)i * CC * CC, bq + i * CC, nullptr, qb, CC, CC, 0);
    cb_attn<<<dim3(7, HH, BN), dim3(32), 0, stream>>>(qb, Kb, Vtb, attnb, scaler);
    // f = attn @ Wf + bf (f32 out)
    cb_gemm<<<dim3(NTOT / 128, CC / 64), dim3(128), 0, stream>>>(
        attnb, Wft + (long)i * CC * CC, bf + i * CC, tmp, nullptr, CC, CC, 0);
    cb_add_ln<<<dim3(NTOT), dim3(32), 0, stream>>>(z, tmp, g1 + i * CC, be1 + i * CC, zb);
    // ffn
    cb_gemm<<<dim3(NTOT / 128, FFD / 64), dim3(128), 0, stream>>>(
        zb, W1t + (long)i * FFD * CC, b1 + i * FFD, nullptr, h1b, CC, FFD, 1);
    cb_gemm<<<dim3(NTOT / 128, CC / 64), dim3(128), 0, stream>>>(
        h1b, W2t + (long)i * CC * FFD, b2 + i * CC, tmp, nullptr, FFD, CC, 0);
    cb_add_ln<<<dim3(NTOT), dim3(32), 0, stream>>>(z, tmp, g2 + i * CC, be2 + i * CC, zb);
  }

  // post FFN
  cb_gemm<<<dim3(NTOT / 128, FFD / 64), dim3(128), 0, stream>>>(
      zb, Wp1t, bp1, nullptr, h1b, CC, FFD, 1);
  cb_gemm<<<dim3(NTOT / 128, CC / 64), dim3(128), 0, stream>>>(
      h1b, Wp2t, bp2, tmp, nullptr, FFD, CC, 0);

  long ncs = (long)NTOT * CC;
  cb_bcast_add<<<dim3((unsigned)((ncs + 255) / 256)), dim3(256), 0, stream>>>(feat, tmp, out);
}
